// OFT_79688823210370
// MI455X (gfx1250) — matrix operand order repaired and checked by reading, not compiled, not run on hardware
//
#include <hip/hip_runtime.h>

typedef __bf16 bf16_t;
typedef __bf16 v16bf __attribute__((ext_vector_type(16)));
typedef __bf16 v8bf  __attribute__((ext_vector_type(8)));
typedef float  v8f   __attribute__((ext_vector_type(8)));
typedef int    vi4   __attribute__((ext_vector_type(4)));

// Problem constants (from reference setup_inputs)
constexpr int Bb = 2, Cc = 128, Hh = 48, Wf = 160;
constexpr int NYc = 5, NDc = 161, NWc = 161;
constexpr int NY1 = 4, Dd = 160, Ww = 160, DW = Dd * Ww;   // 25600
constexpr int Mm = Bb * DW;                                 // 51200
constexpr int Kk = Cc * NY1;                                // 512
constexpr int Nn = 512;
constexpr float EPSf = 1e-6f;

// ---------------------------------------------------------------------------
// CDNA5 async global->LDS copy (ASYNCcnt path), with compile-safe fallback.
// ROCm clang prototype (from diagnostics): params are pointers to <4 x i32>
// in AS(1) (global) / AS(3) (LDS), then imm offset + imm cpol.
// ---------------------------------------------------------------------------
#if defined(__HIP_DEVICE_COMPILE__) && __has_builtin(__builtin_amdgcn_global_load_async_to_lds_b128)
#define HAVE_ASYNC_LDS 1
#else
#define HAVE_ASYNC_LDS 0
#endif

typedef __attribute__((address_space(1))) vi4 g_vi4;
typedef __attribute__((address_space(3))) vi4 l_vi4;

__device__ __forceinline__ void copy_b128_to_lds(const void* gsrc, void* lds) {
#if HAVE_ASYNC_LDS
    __builtin_amdgcn_global_load_async_to_lds_b128(
        (g_vi4*)(unsigned long long)gsrc,
        (l_vi4*)(unsigned)(unsigned long long)lds, 0, 0);
#else
    *(uint4*)lds = *(const uint4*)gsrc;
#endif
}

__device__ __forceinline__ void async_copies_wait() {
#if HAVE_ASYNC_LDS
#if __has_builtin(__builtin_amdgcn_s_wait_asynccnt)
    __builtin_amdgcn_s_wait_asynccnt(0);
#else
    asm volatile("s_wait_asynccnt 0x0" ::: "memory");
#endif
#endif
}

// ---------------------------------------------------------------------------
// Kernel 1a: cumulative sum along W; features [B,C,H,W] -> integral channel-last
// layout [B,H,W,C] so the vox gather is coalesced across channels.
// ---------------------------------------------------------------------------
__global__ void integral_w_kernel(const float* __restrict__ feat,
                                  float* __restrict__ integ) {
    int tid = blockIdx.x * blockDim.x + threadIdx.x;   // B*H*C = 12288
    if (tid >= Bb * Hh * Cc) return;
    int c = tid % Cc;
    int h = (tid / Cc) % Hh;
    int b = tid / (Cc * Hh);
    const float* src = feat + (((size_t)b * Cc + c) * Hh + h) * Wf;
    float* dst = integ + ((size_t)b * Hh + h) * Wf * Cc + c;
    float acc = 0.0f;
    for (int w = 0; w < Wf; ++w) {
        acc += src[w];
        dst[(size_t)w * Cc] = acc;
    }
}

// ---------------------------------------------------------------------------
// Kernel 1b: cumulative sum along H, in place on channel-last integral.
// ---------------------------------------------------------------------------
__global__ void integral_h_kernel(float* __restrict__ integ) {
    int tid = blockIdx.x * blockDim.x + threadIdx.x;   // B*W*C = 40960
    if (tid >= Bb * Wf * Cc) return;
    int c = tid % Cc;
    int w = (tid / Cc) % Wf;
    int b = tid / (Cc * Wf);
    float acc = 0.0f;
    for (int h = 0; h < Hh; ++h) {
        size_t idx = (((size_t)b * Hh + h) * Wf + w) * Cc + c;
        acc += integ[idx];
        integ[idx] = acc;
    }
}

// ---------------------------------------------------------------------------
// Kernel 2: f32 -> bf16 convert + transpose of W_lin: Wt[n][k] = W_lin[k][n].
// Pre-transposing makes every GEMM B-stage a contiguous row copy (b128,
// async-to-LDS capable) instead of scalar ds_store_b16 transposes.
// ---------------------------------------------------------------------------
__global__ void cvt_wt_kernel(const float* __restrict__ src,
                              bf16_t* __restrict__ dst) {
    int i = blockIdx.x * blockDim.x + threadIdx.x;   // over K*N = 262144
    if (i >= Kk * Nn) return;
    int k = i >> 9;      // / 512
    int n = i & 511;
    dst[(size_t)n * Kk + k] = (bf16_t)src[i];
}

// ---------------------------------------------------------------------------
// Projection helper: voxel corner -> clipped normalized image coords.
// ---------------------------------------------------------------------------
__device__ __forceinline__ float2 project_nc(const float* __restrict__ cal,
                                             const float* __restrict__ corners,
                                             int iy, int iz, int ix) {
    const float* p = corners + (((size_t)iy * NDc + iz) * NWc + ix) * 3;
    float X = p[0], Y = p[1], Z = p[2];
    float hx = cal[0] * X + cal[1] * Y + cal[2]  * Z + cal[3];
    float hy = cal[4] * X + cal[5] * Y + cal[6]  * Z + cal[7];
    float hz = cal[8] * X + cal[9] * Y + cal[10] * Z + cal[11];
    float inv = 1.0f / hz;
    float u = hx * inv, v = hy * inv;
    float nx = fminf(fmaxf(2.0f * u / (float)Wf - 1.0f, -1.0f), 1.0f);
    float ny = fminf(fmaxf(2.0f * v / (float)Hh - 1.0f, -1.0f), 1.0f);
    return make_float2(nx, ny);
}

// ---------------------------------------------------------------------------
// Kernel 3: per voxel cell, box-sum the integral image (4 corner samples, each
// bilinear = 16 taps), scaled by visible/area, write bf16 vox row [M,K=512],
// K ordered as c*4 + y (matches reference reshape).
// ---------------------------------------------------------------------------
__global__ __launch_bounds__(128) void vox_kernel(const float* __restrict__ integ,
                                                  const float* __restrict__ calib,
                                                  const float* __restrict__ corners,
                                                  bf16_t* __restrict__ vox) {
    __shared__ float s_w[NY1][16];
    __shared__ int   s_o[NY1][16];

    int r   = blockIdx.x;            // 0..51199, r = b*DW + z*Ww + x
    int b   = r / DW;
    int rem = r - b * DW;
    int z   = rem / Ww;
    int x   = rem - z * Ww;
    int t   = threadIdx.x;

    if (t < NY1) {
        const float* cal = calib + b * 12;
        float2 pa = project_nc(cal, corners, t,     z,     x);
        float2 pb = project_nc(cal, corners, t,     z + 1, x);
        float2 pc = project_nc(cal, corners, t + 1, z + 1, x + 1);
        float2 pd = project_nc(cal, corners, t + 1, z,     x + 1);
        float x0 = fminf(pa.x, pb.x), y0 = fminf(pa.y, pb.y);
        float x1 = fmaxf(pc.x, pd.x), y1 = fmaxf(pc.y, pd.y);
        float area = (x1 - x0) * (y1 - y0) * (0.25f * Hh * Wf) + EPSf;
        float voa  = (area > EPSf) ? (1.0f / area) : 0.0f;   // visible/area

        float px[4] = {x0, x1, x1, x0};   // tl, br, tr, bl
        float py[4] = {y0, y1, y0, y1};
        float sg[4] = {1.0f, 1.0f, -1.0f, -1.0f};
        for (int p = 0; p < 4; ++p) {
            float u = (px[p] + 1.0f) * (0.5f * Wf) - 0.5f;
            float v = (py[p] + 1.0f) * (0.5f * Hh) - 0.5f;
            float uf = floorf(u), vf = floorf(v);
            float fx = u - uf, fy = v - vf;
            int xi = (int)uf, yi = (int)vf;
            float wq[4] = {(1.f - fx) * (1.f - fy), fx * (1.f - fy),
                           (1.f - fx) * fy,         fx * fy};
            int dxs[4] = {0, 1, 0, 1};
            int dys[4] = {0, 0, 1, 1};
            for (int q = 0; q < 4; ++q) {
                int xx = xi + dxs[q], yy = yi + dys[q];
                bool valid = (xx >= 0) && (xx < Wf) && (yy >= 0) && (yy < Hh);
                int xc = min(max(xx, 0), Wf - 1);
                int yc = min(max(yy, 0), Hh - 1);
                s_w[t][p * 4 + q] = valid ? (wq[q] * sg[p] * voa) : 0.0f;
                s_o[t][p * 4 + q] = (yc * Wf + xc) * Cc;   // channel-last plane offset
            }
        }
    }
    __syncthreads();

    int c = t;                                   // 0..127
    const float* Ib = integ + (size_t)b * Hh * Wf * Cc + c;
    bf16_t* vrow = vox + (size_t)r * Kk + c * NY1;
    #pragma unroll
    for (int y = 0; y < NY1; ++y) {
        float acc = 0.0f;
        #pragma unroll
        for (int j = 0; j < 16; ++j)
            acc += s_w[y][j] * Ib[s_o[y][j]];
        vrow[y] = (bf16_t)acc;
    }
}

// ---------------------------------------------------------------------------
// Kernel 4: GEMM  out = relu(vox[M,K] * W[K,N] + bias), fused transposed
// scatter out[b, n, w, d].  bf16 WMMA 16x16x32, f32 accumulate.
//
// Block tile: 64 M x 512 N (full N extent -> vox is read exactly once).
// 256 threads = 8 waves: wm = wv&3 picks the 16-row M strip, ng = wv>>2 picks
// the 256-wide N half; each wave owns 16 N sub-tiles -> 16 WMMA per K-step
// off a single A fragment (A-frag reuse).
// LDS: As[64][32] (m,k); Bs[512][32] from pre-transposed Wt[n][k] so all
// staging is contiguous b128 rows, issued as GLOBAL_LOAD_ASYNC_TO_LDS_B128.
// ---------------------------------------------------------------------------
__global__ __launch_bounds__(256) void gemm_kernel(const bf16_t* __restrict__ A,
                                                   const bf16_t* __restrict__ Wt,
                                                   const float* __restrict__ bias,
                                                   float* __restrict__ out) {
    __shared__ __attribute__((aligned(16))) bf16_t As[64 * 32];    //  4 KB
    __shared__ __attribute__((aligned(32))) bf16_t Bs[Nn * 32];    // 32 KB

    const int tid  = threadIdx.x;
    const int lane = tid & 31;
    const int wv   = tid >> 5;        // wave 0..7
    const int wm   = wv & 3;          // M sub-tile 0..3
    const int ng   = wv >> 2;         // N half 0..1
    const int n_base = ng * 256;
    const int Moff = blockIdx.x * 64;

    const int hl  = lane >> 4;        // lane half
    const int l16 = lane & 15;

    v8f acc[16];
    #pragma unroll
    for (int j = 0; j < 16; ++j)
        acc[j] = (v8f){0.f, 0.f, 0.f, 0.f, 0.f, 0.f, 0.f, 0.f};

    // A staging: thread -> (row 0..63, 16B chunk 0..3), one b128 each.
    const int arow = tid >> 2;
    const int aj   = tid & 3;
    const int m_local = wm * 16 + l16;

    #pragma unroll 1
    for (int ks = 0; ks < Kk; ks += 32) {
        if (ks + 32 < Kk)   // speculative prefetch of next A tile
            __builtin_prefetch(A + (size_t)(Moff + arow) * Kk + ks + 32 + aj * 8, 0, 1);

        // stage A tile 64x32 (one async b128 per thread)
        copy_b128_to_lds(A + (size_t)(Moff + arow) * Kk + ks + aj * 8,
                         &As[arow * 32 + aj * 8]);
        // stage B tile 512x32 from Wt[n][k] (eight async b128 per thread)
        #pragma unroll
        for (int i = 0; i < 8; ++i) {
            int id  = i * 256 + tid;      // 0..2047
            int row = id >> 2;            // n row 0..511
            int ch  = id & 3;             // 16B chunk in the 64B row slice
            copy_b128_to_lds(Wt + (size_t)row * Kk + ks + ch * 8,
                             &Bs[row * 32 + ch * 8]);
        }
        async_copies_wait();
        __syncthreads();

        // A fragment (ISA 16-bit A layout: K groups of 8 interleave by lane half)
        v8bf alo = *(const v8bf*)(&As[m_local * 32 + hl * 8]);
        v8bf ahi = *(const v8bf*)(&As[m_local * 32 + 16 + hl * 8]);
        v16bf afrag = __builtin_shufflevector(alo, ahi,
            0, 1, 2, 3, 4, 5, 6, 7, 8, 9, 10, 11, 12, 13, 14, 15);

        // 16 WMMA per K-step re-using the A fragment
        #pragma unroll
        for (int j = 0; j < 16; ++j) {
            const int n_local = n_base + j * 16 + l16;
            v8bf blo = *(const v8bf*)(&Bs[n_local * 32 + hl * 8]);
            v8bf bhi = *(const v8bf*)(&Bs[n_local * 32 + 16 + hl * 8]);
            v16bf bfrag = __builtin_shufflevector(blo, bhi,
                0, 1, 2, 3, 4, 5, 6, 7, 8, 9, 10, 11, 12, 13, 14, 15);
            acc[j] = __builtin_amdgcn_wmma_f32_16x16x32_bf16(
                false, afrag, false, bfrag, (short)0, acc[j], false, false);
        }
        __syncthreads();
    }

    // Epilogue: bias + relu + transposed scatter out[b, n, w, d].
    // out idx = b*Nn*DW + n*DW + w*Dd + d  with m = b*DW + d*Ww + w.
    size_t obase[8];
    #pragma unroll
    for (int rr = 0; rr < 8; ++rr) {
        int m    = Moff + wm * 16 + rr + hl * 8;
        int b    = m / DW;
        int rem  = m - b * DW;
        int dcol = rem / Ww;
        int wcol = rem - dcol * Ww;
        obase[rr] = (size_t)b * Nn * DW + (size_t)wcol * Dd + dcol;
    }
    #pragma unroll
    for (int j = 0; j < 16; ++j) {
        const int n_g = n_base + j * 16 + l16;
        const float bv = bias[n_g];
        #pragma unroll
        for (int rr = 0; rr < 8; ++rr) {
            float v = acc[j][rr] + bv;
            v = v > 0.0f ? v : 0.0f;
            out[obase[rr] + (size_t)n_g * DW] = v;
        }
    }
}

// ---------------------------------------------------------------------------
extern "C" void kernel_launch(void* const* d_in, const int* in_sizes, int n_in,
                              void* d_out, int out_size, void* d_ws, size_t ws_size,
                              hipStream_t stream) {
    (void)in_sizes; (void)n_in; (void)out_size; (void)ws_size;

    const float* features = (const float*)d_in[0];
    const float* calib    = (const float*)d_in[1];
    const float* corners  = (const float*)d_in[2];
    const float* W_lin    = (const float*)d_in[3];
    const float* b_lin    = (const float*)d_in[4];
    float* out = (float*)d_out;
    char*  ws  = (char*)d_ws;

    // Workspace layout
    float*  integ = (float*)ws;                                  // 7.86 MB
    size_t  off   = (size_t)Bb * Hh * Wf * Cc * sizeof(float);
    bf16_t* vox   = (bf16_t*)(ws + off);                         // 52.4 MB
    off += (size_t)Mm * Kk * sizeof(bf16_t);
    bf16_t* Wt    = (bf16_t*)(ws + off);                         // 0.5 MB (transposed)

    integral_w_kernel<<<(Bb * Hh * Cc + 255) / 256, 256, 0, stream>>>(features, integ);
    integral_h_kernel<<<(Bb * Wf * Cc + 255) / 256, 256, 0, stream>>>(integ);
    cvt_wt_kernel<<<(Kk * Nn + 255) / 256, 256, 0, stream>>>(W_lin, Wt);
    vox_kernel<<<Mm, 128, 0, stream>>>(integ, calib, corners, vox);

    gemm_kernel<<<Mm / 64, 256, 0, stream>>>(vox, Wt, b_lin, out);
}
